// DNNLSTMWithAttention_72945724555345
// MI455X (gfx1250) — matrix operand order repaired and checked by reading, not compiled, not run on hardware
//
#include <hip/hip_runtime.h>
#include <hip/hip_bf16.h>
#include <math.h>

// ---------------------------------------------------------------------------
// Fused MLP + LSTM-cell + (degenerate) attention + FC for MI455X (gfx1250).
// All GEMMs on v_wmma_f32_16x16x32_bf16. Weights pre-swizzled to bf16 WMMA
// B-fragment layout (one contiguous 32B load per lane per k-step). x staged
// into LDS once (async DMA, ASYNCcnt-tracked), 32 batch rows per wave with
// two accumulator tiles sharing every B fragment -> halves L2 weight traffic
// and keeps 6 independent WMMA chains in flight in the gates loop.
// HBM floor: read x (268MB) + write out (67MB) ~ 14us at 23.3 TB/s.
// ---------------------------------------------------------------------------

typedef __attribute__((ext_vector_type(16))) __bf16 v16bf;
typedef __attribute__((ext_vector_type(8)))  __bf16 v8bf;
typedef __attribute__((ext_vector_type(8)))  float  v8f;
typedef __attribute__((ext_vector_type(4)))  int    v4i;

// async-to-LDS pointer types: AS1 (global) and AS3 (LDS) v4i*
typedef __attribute__((address_space(1))) v4i* g_v4i_p;
typedef __attribute__((address_space(3))) v4i* l_v4i_p;

#define B_TOT   131072
#define IN_D    512
#define OUT_D   128
#define XC_P    648   // padded LDS row stride (bank-conflict free, 16B aligned)
#define D1_P    136
#define D2_P    72
#define NWAVE   2     // independent waves per block
#define ROWS_W  32    // batch rows per wave (2 WMMA M-tiles)

// per-wave dynamic LDS layout (bytes)
#define XC_OFF   0
#define D1_OFF   (XC_OFF + ROWS_W * XC_P * 2)          // 41472
#define D2_OFF   (D1_OFF + ROWS_W * D1_P * 2)          // +8704
#define STG_OFF  (D2_OFF + ROWS_W * D2_P * 2)          // +4608
#define WAVE_BYTES (STG_OFF + 2 * IN_D * 4)            // +4096 = 58880
#define LDS_TOTAL  (NWAVE * WAVE_BYTES)                // 117760

// workspace element offsets (bf16 elements)
#define OFF_W1   0L
#define OFF_W2   (OFF_W1 + 128L * 512L)
#define OFF_W3   (OFF_W2 + 64L * 128L)
#define OFF_WIH  (OFF_W3 + 128L * 64L)
#define OFF_WFC  (OFF_WIH + 512L * 640L)

#if __has_builtin(__builtin_amdgcn_global_load_async_to_lds_b128) && \
    __has_builtin(__builtin_amdgcn_s_wait_asynccnt)
#define HAVE_ASYNC_LDS 1
#else
#define HAVE_ASYNC_LDS 0
#endif

static __device__ __forceinline__ unsigned short f2bfu(float f) {
  unsigned u = __builtin_bit_cast(unsigned, f);
  u += 0x7FFFu + ((u >> 16) & 1u);            // round-to-nearest-even
  return (unsigned short)(u >> 16);
}

static __device__ __forceinline__ v8f wmma_bf16(v16bf a, v16bf b, v8f c) {
  // (neg_a, A, neg_b, B, c_mod, C, reuse_a, reuse_b)
  return __builtin_amdgcn_wmma_f32_16x16x32_bf16(false, a, false, b, (short)0,
                                                 c, false, false);
}

// A-matrix fragment (16x32 bf16) from LDS, contiguous k order (same as swizzle_w_kernel):
// lanes 0-15 (half=0): row=lane, K = k0+0..15
// lanes 16-31 (half=1): row=lane-16, K = k0+16..31
static __device__ __forceinline__ v16bf a_frag(const unsigned short* rowbase,
                                               int k0, int half) {
  v8bf lo = *(const v8bf*)(rowbase + k0 + half * 16);
  v8bf hi = *(const v8bf*)(rowbase + k0 + half * 16 + 8);
  return __builtin_shufflevector(lo, hi, 0, 1, 2, 3, 4, 5, 6, 7,
                                 8, 9, 10, 11, 12, 13, 14, 15);
}

// Pre-swizzled B fragment: 16 consecutive bf16 (32B) per lane per k-step.
static __device__ __forceinline__ v16bf b_frag(const unsigned short* w, long frag) {
  return *(const v16bf*)(w + (frag << 4));
}

static __device__ __forceinline__ float sigm(float x) {
  return 1.0f / (1.0f + expf(-x));
}

// ---------------------------------------------------------------------------
// Weight prep: fp32 row-major (N,K) -> bf16 in WMMA B-fragment order.
//   dst[(tile*32 + lane)*16 + e] = W[nt*16 + (lane&15)][kt*32 + (lane>>4)*16 + e]
// ---------------------------------------------------------------------------
__global__ __launch_bounds__(256) void swizzle_w_kernel(
    const float* __restrict__ src, unsigned short* __restrict__ dst,
    int N, int K, long total) {
  long t = (long)blockIdx.x * 256 + threadIdx.x;
  if (t >= total) return;
  int  nk   = K >> 5;
  long tile = t >> 9;          // 512 elements per 16x32 tile
  int  r    = (int)(t & 511);
  int  lane = r >> 4;
  int  e    = r & 15;
  long kt   = tile % nk;
  long nt   = tile / nk;
  long n    = nt * 16 + (lane & 15);
  long k    = kt * 32 + (lane >> 4) * 16 + e;
  dst[t] = f2bfu(src[n * (long)K + k]);
}

// ---------------------------------------------------------------------------
// Fused kernel: each wave owns a 32-row batch tile end-to-end.
// ---------------------------------------------------------------------------
__global__ __launch_bounds__(NWAVE * 32) void fused_mlp_lstm_attn(
    const float* __restrict__ x,
    const float* __restrict__ fc1_b, const float* __restrict__ fc2_b,
    const float* __restrict__ fc3_b, const float* __restrict__ b_ih,
    const float* __restrict__ b_hh, const float* __restrict__ fc_b,
    const unsigned short* __restrict__ ws,
    float* __restrict__ out) {
  extern __shared__ __align__(16) char dyn_smem[];

  const int  wave = threadIdx.x >> 5;
  const int  lane = threadIdx.x & 31;
  const int  half = lane >> 4;
  const int  lm   = lane & 15;
  const long row0 = ((long)blockIdx.x * NWAVE + wave) * ROWS_W;

  char* wbase = dyn_smem + wave * WAVE_BYTES;
  unsigned short* XCw = (unsigned short*)(wbase + XC_OFF);
  unsigned short* D1w = (unsigned short*)(wbase + D1_OFF);
  unsigned short* D2w = (unsigned short*)(wbase + D2_OFF);
  float*          stg = (float*)(wbase + STG_OFF);   // 2 x 512 fp32 buffers

  const unsigned short* w1  = ws + OFF_W1;
  const unsigned short* w2  = ws + OFF_W2;
  const unsigned short* w3  = ws + OFF_W3;
  const unsigned short* wih = ws + OFF_WIH;
  const unsigned short* wfc = ws + OFF_WFC;

  // ---- Stage x (32 x 512 fp32) into LDS as bf16 ---------------------------
  auto convert_row = [&](int r, const float* sb) {
#pragma unroll
    for (int i = 0; i < 4; ++i) {
      float4 v = *(const float4*)(sb + (lane + 32 * i) * 4);
      ushort4 pk = make_ushort4(f2bfu(v.x), f2bfu(v.y), f2bfu(v.z), f2bfu(v.w));
      *(ushort4*)(XCw + r * XC_P + (lane + 32 * i) * 4) = pk;
    }
  };

#if HAVE_ASYNC_LDS
  // Async DMA (ASYNCcnt-tracked) double-buffered staging: row r+1 in flight
  // while row r is converted fp32 -> bf16.
  auto issue_row = [&](long row, int buf) {
    const float* src = x + row * IN_D;
    float*       dst = stg + buf * IN_D;
#pragma unroll
    for (int i = 0; i < 4; ++i) {
      __builtin_amdgcn_global_load_async_to_lds_b128(
          (g_v4i_p)(src + (lane + 32 * i) * 4),
          (l_v4i_p)(dst + (lane + 32 * i) * 4),
          0, 0);
    }
  };
  issue_row(row0, 0);
  for (int r = 0; r < ROWS_W - 1; ++r) {
    issue_row(row0 + r + 1, (r + 1) & 1);
    __builtin_amdgcn_s_wait_asynccnt(4);   // row r's 4 DMAs complete
    convert_row(r, stg + (r & 1) * IN_D);
  }
  __builtin_amdgcn_s_wait_asynccnt(0);
  convert_row(ROWS_W - 1, stg + ((ROWS_W - 1) & 1) * IN_D);
#else
  for (int r = 0; r < ROWS_W; ++r) {
    const float4* src = (const float4*)(x + (row0 + r) * IN_D);
#pragma unroll
    for (int i = 0; i < 4; ++i) {
      float4 v = src[lane + 32 * i];
      ushort4 pk = make_ushort4(f2bfu(v.x), f2bfu(v.y), f2bfu(v.z), f2bfu(v.w));
      *(ushort4*)(XCw + r * XC_P + (lane + 32 * i) * 4) = pk;
    }
  }
#endif

  // per-tile A row bases (tile 0: rows 0-15, tile 1: rows 16-31)
  const unsigned short* Axc0 = XCw + lm * XC_P;
  const unsigned short* Axc1 = XCw + (16 + lm) * XC_P;
  const unsigned short* Ad1_0 = D1w + lm * D1_P;
  const unsigned short* Ad1_1 = D1w + (16 + lm) * D1_P;
  const unsigned short* Ad2_0 = D2w + lm * D2_P;
  const unsigned short* Ad2_1 = D2w + (16 + lm) * D2_P;

  // ---- fc1: d1 = relu(x @ fc1_w^T + b1), 32x128, K=512 --------------------
  for (int nt = 0; nt < 8; ++nt) {
    v8f a0 = {}, a1 = {};
#pragma unroll 4
    for (int kt = 0; kt < 16; ++kt) {
      v16bf b = b_frag(w1, (long)(nt * 16 + kt) * 32 + lane);
      a0 = wmma_bf16(a_frag(Axc0, kt * 32, half), b, a0);
      a1 = wmma_bf16(a_frag(Axc1, kt * 32, half), b, a1);
    }
    int   col  = nt * 16 + lm;
    float bias = fc1_b[col];
#pragma unroll
    for (int j = 0; j < 8; ++j) {
      float v0 = a0[j] + bias; v0 = v0 > 0.f ? v0 : 0.f;
      float v1 = a1[j] + bias; v1 = v1 > 0.f ? v1 : 0.f;
      D1w[(j + 8 * half) * D1_P + col]        = f2bfu(v0);
      D1w[(16 + j + 8 * half) * D1_P + col]   = f2bfu(v1);
    }
  }

  // ---- fc2: d2 = relu(d1 @ fc2_w^T + b2), 32x64, K=128 --------------------
  for (int nt = 0; nt < 4; ++nt) {
    v8f a0 = {}, a1 = {};
#pragma unroll
    for (int kt = 0; kt < 4; ++kt) {
      v16bf b = b_frag(w2, (long)(nt * 4 + kt) * 32 + lane);
      a0 = wmma_bf16(a_frag(Ad1_0, kt * 32, half), b, a0);
      a1 = wmma_bf16(a_frag(Ad1_1, kt * 32, half), b, a1);
    }
    int   col  = nt * 16 + lm;
    float bias = fc2_b[col];
#pragma unroll
    for (int j = 0; j < 8; ++j) {
      float v0 = a0[j] + bias; v0 = v0 > 0.f ? v0 : 0.f;
      float v1 = a1[j] + bias; v1 = v1 > 0.f ? v1 : 0.f;
      D2w[(j + 8 * half) * D2_P + col]        = f2bfu(v0);
      D2w[(16 + j + 8 * half) * D2_P + col]   = f2bfu(v1);
    }
  }

  // ---- fc3: d3 = d2 @ fc3_w^T + b3 -> xc columns 512..639 -----------------
  for (int nt = 0; nt < 8; ++nt) {
    v8f a0 = {}, a1 = {};
#pragma unroll
    for (int kt = 0; kt < 2; ++kt) {
      v16bf b = b_frag(w3, (long)(nt * 2 + kt) * 32 + lane);
      a0 = wmma_bf16(a_frag(Ad2_0, kt * 32, half), b, a0);
      a1 = wmma_bf16(a_frag(Ad2_1, kt * 32, half), b, a1);
    }
    int   col  = nt * 16 + lm;
    float bias = fc3_b[col];
#pragma unroll
    for (int j = 0; j < 8; ++j) {
      XCw[(j + 8 * half) * XC_P + 512 + col]      = f2bfu(a0[j] + bias);
      XCw[(16 + j + 8 * half) * XC_P + 512 + col] = f2bfu(a1[j] + bias);
    }
  }

  // ---- gates = xc @ W_ih^T + b_ih + b_hh; only i/g/o needed (f is dead) ---
  // W_ih is (512,640): i -> ntiles 0..7, g -> 16..23, o -> 24..31 (nk=20).
  // 6 independent accumulator chains per iteration; each B fragment feeds
  // two M-tiles (halves L2 weight traffic per batch row).
  for (int nh = 0; nh < 8; ++nh) {
    v8f ai0 = {}, ag0 = {}, ao0 = {};
    v8f ai1 = {}, ag1 = {}, ao1 = {};
#pragma unroll 2
    for (int kt = 0; kt < 20; ++kt) {
      v16bf a0 = a_frag(Axc0, kt * 32, half);
      v16bf a1 = a_frag(Axc1, kt * 32, half);
      v16bf bi = b_frag(wih, (long)((nh)*20 + kt) * 32 + lane);
      v16bf bg = b_frag(wih, (long)((nh + 16) * 20 + kt) * 32 + lane);
      v16bf bo = b_frag(wih, (long)((nh + 24) * 20 + kt) * 32 + lane);
      ai0 = wmma_bf16(a0, bi, ai0);  ai1 = wmma_bf16(a1, bi, ai1);
      ag0 = wmma_bf16(a0, bg, ag0);  ag1 = wmma_bf16(a1, bg, ag1);
      ao0 = wmma_bf16(a0, bo, ao0);  ao1 = wmma_bf16(a1, bo, ao1);
    }
    int   col = nh * 16 + lm;
    float bi  = b_ih[col] + b_hh[col];
    float bg  = b_ih[256 + col] + b_hh[256 + col];
    float bo  = b_ih[384 + col] + b_hh[384 + col];
#pragma unroll
    for (int j = 0; j < 8; ++j) {
      float iv0 = sigm(ai0[j] + bi), iv1 = sigm(ai1[j] + bi);
      float gv0 = tanhf(ag0[j] + bg), gv1 = tanhf(ag1[j] + bg);
      float ov0 = sigm(ao0[j] + bo), ov1 = sigm(ao1[j] + bo);
      float h0 = ov0 * tanhf(iv0 * gv0);
      float h1 = ov1 * tanhf(iv1 * gv1);
      // seq-len-1 softmax == 1 -> ctx == h; store h (reuse d1 buffer)
      D1w[(j + 8 * half) * D1_P + col]      = f2bfu(h0);
      D1w[(16 + j + 8 * half) * D1_P + col] = f2bfu(h1);
    }
  }

  // ---- out = h @ fc_w^T + fc_b, 32x128, K=128 -----------------------------
  for (int nt = 0; nt < 8; ++nt) {
    v8f a0 = {}, a1 = {};
#pragma unroll
    for (int kt = 0; kt < 4; ++kt) {
      v16bf b = b_frag(wfc, (long)(nt * 4 + kt) * 32 + lane);
      a0 = wmma_bf16(a_frag(Ad1_0, kt * 32, half), b, a0);
      a1 = wmma_bf16(a_frag(Ad1_1, kt * 32, half), b, a1);
    }
    int   col  = nt * 16 + lm;
    float bias = fc_b[col];
#pragma unroll
    for (int j = 0; j < 8; ++j) {
      out[(row0 + j + 8 * half) * OUT_D + col]      = a0[j] + bias;
      out[(row0 + 16 + j + 8 * half) * OUT_D + col] = a1[j] + bias;
    }
  }
}

// ---------------------------------------------------------------------------
extern "C" void kernel_launch(void* const* d_in, const int* in_sizes, int n_in,
                              void* d_out, int out_size, void* d_ws,
                              size_t ws_size, hipStream_t stream) {
  const float* x     = (const float*)d_in[0];
  const float* fc1_w = (const float*)d_in[1];
  const float* fc1_b = (const float*)d_in[2];
  const float* fc2_w = (const float*)d_in[3];
  const float* fc2_b = (const float*)d_in[4];
  const float* fc3_w = (const float*)d_in[5];
  const float* fc3_b = (const float*)d_in[6];
  const float* W_ih  = (const float*)d_in[7];
  // d_in[8] = W_hh   : unused by the reference (no recurrent state)
  const float* b_ih  = (const float*)d_in[9];
  const float* b_hh  = (const float*)d_in[10];
  // d_in[11] = attn_w, d_in[12] = attn_b : cancel in length-1 softmax
  const float* fc_w  = (const float*)d_in[13];
  const float* fc_b  = (const float*)d_in[14];

  unsigned short* ws  = (unsigned short*)d_ws;
  float*          out = (float*)d_out;

  // Prep: fp32 -> bf16, swizzled to WMMA B-fragment order (~0.85MB total).
  auto launch_sw = [&](const float* s, long off, int N, int K) {
    long total  = (long)N * K;
    int  blocks = (int)((total + 255) / 256);
    swizzle_w_kernel<<<blocks, 256, 0, stream>>>(s, ws + off, N, K, total);
  };
  launch_sw(fc1_w, OFF_W1, 128, 512);
  launch_sw(fc2_w, OFF_W2, 64, 128);
  launch_sw(fc3_w, OFF_W3, 128, 64);
  launch_sw(W_ih, OFF_WIH, 512, 640);
  launch_sw(fc_w, OFF_WFC, 128, 128);

  int blocks = B_TOT / (ROWS_W * NWAVE);  // 2048 blocks x 64 threads
  fused_mlp_lstm_attn<<<blocks, NWAVE * 32, LDS_TOTAL, stream>>>(
      x, fc1_b, fc2_b, fc3_b, b_ih, b_hh, fc_b, ws, out);
}
